// QKVAttention_55336358643304
// MI455X (gfx1250) — matrix operand order repaired and checked by reading, not compiled, not run on hardware
//
#include <hip/hip_runtime.h>

typedef _Float16 h16;
typedef __attribute__((ext_vector_type(16))) _Float16 v16h;
typedef __attribute__((ext_vector_type(8)))  _Float16 v8h;
typedef __attribute__((ext_vector_type(4)))  _Float16 v4h;
typedef __attribute__((ext_vector_type(2)))  _Float16 v2h;
typedef __attribute__((ext_vector_type(8)))  float    v8f;

#define T_LEN 1024
#define BM 64          // query tile
#define BN 64          // key tile
#define LDP 72         // padded LDS row (h16 elements); 144B rows keep 16B alignment

// v_cvt_pk_rtz_f16_f32: builtin returns an __fp16 vector; bit-cast to _Float16 vector
__device__ __forceinline__ v2h pkrtz(float a, float b) {
  return __builtin_bit_cast(v2h, __builtin_amdgcn_cvt_pkrtz(a, b));
}

// VALU DPP lane exchange (stays within 16-lane rows); no LDS-pipe traffic, and
// mov_dpp (no tied 'old') lets the backend fold the permute into the consumer.
template <int CTRL>
__device__ __forceinline__ float dppf(float x) {
  int xi = __builtin_bit_cast(int, x);
  return __builtin_bit_cast(float,
      __builtin_amdgcn_mov_dpp(xi, CTRL, 0xf, 0xf, true));
}
// XOR masks {1,3,7,15} are independent -> full 16-lane butterfly:
#define DPP_XOR1  0xB1   // quad_perm(1,0,3,2)
#define DPP_XOR3  0x1B   // quad_perm(3,2,1,0)
#define DPP_XOR7  0x141  // row_half_mirror
#define DPP_XOR15 0x140  // row_mirror

__device__ __forceinline__ float rowMax16(float x) {
  x = fmaxf(x, dppf<DPP_XOR1>(x));
  x = fmaxf(x, dppf<DPP_XOR3>(x));
  x = fmaxf(x, dppf<DPP_XOR7>(x));
  x = fmaxf(x, dppf<DPP_XOR15>(x));
  return x;
}
__device__ __forceinline__ float rowSum16(float x) {
  x += dppf<DPP_XOR1>(x);
  x += dppf<DPP_XOR3>(x);
  x += dppf<DPP_XOR7>(x);
  x += dppf<DPP_XOR15>(x);
  return x;
}

__device__ __forceinline__ v16h cat8(v8h lo, v8h hi) {
  return __builtin_shufflevector(lo, hi, 0,1,2,3,4,5,6,7,8,9,10,11,12,13,14,15);
}

// B-fragment (32x16, K x N): n=lane&15; k = kg+s (s<8), 16+kg+(s-8) (s>=8), kg=(lane>>4)*8.
__device__ __forceinline__ v16h ldB(const h16* p) {
  v8h lo = *(const v8h*)(p);
  v8h hi = *(const v8h*)(p + 16);
  return cat8(lo, hi);
}

// A-fragment (16x32, M x K): lane row = lane&15; kg=(lane>>4)*8.
__device__ __forceinline__ v16h ldA(const h16* rowp, int kk, int kg) {
  v8h lo = *(const v8h*)(rowp + kk*32 + kg);
  v8h hi = *(const v8h*)(rowp + kk*32 + 16 + kg);
  return cat8(lo, hi);
}

// pack 8 consecutive f32 (two float4) into 8 f16 via v_cvt_pk_rtz_f16_f32
__device__ __forceinline__ v8h packRow8(float4 a, float4 b) {
  v2h p0 = pkrtz(a.x, a.y);
  v2h p1 = pkrtz(a.z, a.w);
  v2h p2 = pkrtz(b.x, b.y);
  v2h p3 = pkrtz(b.z, b.w);
  v4h lo = __builtin_shufflevector(p0, p1, 0,1,2,3);
  v4h hi = __builtin_shufflevector(p2, p3, 0,1,2,3);
  return __builtin_shufflevector(lo, hi, 0,1,2,3,4,5,6,7);
}

__global__ __launch_bounds__(128) void qkv_flash_attn(
    const float* __restrict__ qkv,
    const unsigned char* __restrict__ mask,
    const float* __restrict__ qk_bias,
    float* __restrict__ out)
{
  __shared__ h16 Ql[BM][LDP];                      // [t][c]  (transposed in)
  __shared__ h16 Kl[2][BN][LDP];                   // [s][c]  (transposed in), ping-pong
  __shared__ h16 Vl[2][64][LDP];                   // [c][s]  (direct),        ping-pong
  __shared__ h16 Pl[BM][LDP];                      // [t][s]  wave-private rows
  __shared__ __align__(16) unsigned char Msk[2][BM][BN];

  const int bh = blockIdx.y;                       // 0..127 = b*16 + h
  const int qt = blockIdx.x;                       // 0..15
  const int b  = bh >> 4;
  const int hh = bh & 15;
  const int mb = bh & 7;                           // tile(mask,(16,1,1)) => mask[(b*16+h)%8] == mask[h&7]
  const int t0 = qt * BM;

  const int tid  = threadIdx.x;
  const int lane = tid & 31;
  const int nl   = lane & 15;
  const int hlf  = lane >> 4;
  const int wave = tid >> 5;
  const int rW   = wave * 16;
  const int kg   = hlf * 8;

  // qk_bias is a scalar added to EVERY score; softmax(S + b) == softmax(S),
  // so it cancels exactly and is not applied (bit-identical for b == 0).
  (void)qk_bias;

  // exp2 domain: fold scale^2 * log2(e) = 0.125 * 1.4426950 into Q.
  const float QS = 0.125f * 1.44269504088896340736f;

  const size_t base = (size_t)b * (3072u * 1024u) + (size_t)hh * (192u * 1024u);

  // cooperative loader geometry: thread -> (column pair c0, 16-wide s segment)
  const int cp   = tid >> 2;                       // 0..31
  const int c0   = cp * 2;                         // even column
  const int sseg = (tid & 3) * 16;                 // 0,16,32,48
  const float* gk0 = qkv + base + (size_t)(64  + c0) * T_LEN + sseg;
  const float* gv0 = qkv + base + (size_t)(128 + c0) * T_LEN + sseg;
  const int tr = tid >> 1;                         // mask loader: row, 32B half
  const int mo = (tid & 1) * 32;
  const unsigned char* gmb = mask + (size_t)mb * T_LEN * T_LEN
                           + (size_t)(t0 + tr) * T_LEN + mo;

  // ---- f16 register staging for double buffering (40 VGPRs, not 72) ----
  v2h   kst[16];      // K: (c0,c0+1) pair for 16 s values
  v8h   vst[4];       // V: rows c0,c0+1 x two 8-wide s chunks
  uint4 mr[2];

  auto loadKVM = [&](int s0) {
#pragma unroll
    for (int i = 0; i < 4; ++i) {
      float4 k0 = *(const float4*)(gk0 + s0 + i * 4);
      float4 k1 = *(const float4*)(gk0 + T_LEN + s0 + i * 4);
      kst[4*i+0] = pkrtz(k0.x, k1.x);
      kst[4*i+1] = pkrtz(k0.y, k1.y);
      kst[4*i+2] = pkrtz(k0.z, k1.z);
      kst[4*i+3] = pkrtz(k0.w, k1.w);
    }
    float4 a0 = *(const float4*)(gv0 + s0 + 0);
    float4 a1 = *(const float4*)(gv0 + s0 + 4);
    float4 a2 = *(const float4*)(gv0 + s0 + 8);
    float4 a3 = *(const float4*)(gv0 + s0 + 12);
    float4 b0 = *(const float4*)(gv0 + T_LEN + s0 + 0);
    float4 b1 = *(const float4*)(gv0 + T_LEN + s0 + 4);
    float4 b2 = *(const float4*)(gv0 + T_LEN + s0 + 8);
    float4 b3 = *(const float4*)(gv0 + T_LEN + s0 + 12);
    vst[0] = packRow8(a0, a1);
    vst[1] = packRow8(a2, a3);
    vst[2] = packRow8(b0, b1);
    vst[3] = packRow8(b2, b3);
    const uint4* gm = (const uint4*)(gmb + s0);
    mr[0] = gm[0];
    mr[1] = gm[1];
  };
  auto storeKVM = [&](int buf) {
#pragma unroll
    for (int i = 0; i < 16; ++i)                   // K transpose: packed column pair
      *(v2h*)&Kl[buf][sseg + i][c0] = kst[i];
    *(v8h*)&Vl[buf][c0    ][sseg]     = vst[0];    // V direct rows
    *(v8h*)&Vl[buf][c0    ][sseg + 8] = vst[1];
    *(v8h*)&Vl[buf][c0 + 1][sseg]     = vst[2];
    *(v8h*)&Vl[buf][c0 + 1][sseg + 8] = vst[3];
    *(uint4*)&Msk[buf][tr][mo]      = mr[0];
    *(uint4*)&Msk[buf][tr][mo + 16] = mr[1];
  };

  // ---- prologue: tile 0 + Q tile ----
  loadKVM(0);
  {
    const float* gq0 = qkv + base + (size_t)c0 * T_LEN + t0 + sseg;
#pragma unroll
    for (int i = 0; i < 4; ++i) {
      float4 a = *(const float4*)(gq0 + i * 4);
      float4 d = *(const float4*)(gq0 + T_LEN + i * 4);
      int tt = sseg + i * 4;
      *(v2h*)&Ql[tt+0][c0] = pkrtz(a.x * QS, d.x * QS);
      *(v2h*)&Ql[tt+1][c0] = pkrtz(a.y * QS, d.y * QS);
      *(v2h*)&Ql[tt+2][c0] = pkrtz(a.z * QS, d.z * QS);
      *(v2h*)&Ql[tt+3][c0] = pkrtz(a.w * QS, d.w * QS);
    }
  }
  storeKVM(0);
  __syncthreads();

  const h16* qrow = &Ql[rW + nl][0];
  v16h qa0 = ldA(qrow, 0, kg);
  v16h qa1 = ldA(qrow, 1, kg);

  float m[8], l[8];
  v8f acc[4];
#pragma unroll
  for (int j = 0; j < 8; ++j) { m[j] = -1e30f; l[j] = 0.f; }
#pragma unroll
  for (int f = 0; f < 4; ++f)
#pragma unroll
    for (int j = 0; j < 8; ++j) acc[f][j] = 0.f;

#pragma unroll 2
  for (int it = 0; it < 16; ++it) {
    const int cur = it & 1;
    if (it < 15) loadKVM((it + 1) * BN);           // prefetch next tile into VGPRs

    // ---- S2 = (Q * 0.125*log2e) K^T : fragment f, lane nl owns s = 4*nl + f ----
    v8f sf[4];
#pragma unroll
    for (int f = 0; f < 4; ++f) {
      const h16* krow = &Kl[cur][4 * nl + f][hlf * 8];
      v16h bk0 = ldB(krow);
      v16h bk1 = ldB(krow + 32);
      v8f cf;
#pragma unroll
      for (int j = 0; j < 8; ++j) cf[j] = 0.f;
      cf = __builtin_amdgcn_wmma_f32_16x16x32_f16(false, qa0, false, bk0, (short)0, cf, false, false);
      cf = __builtin_amdgcn_wmma_f32_16x16x32_f16(false, qa1, false, bk1, (short)0, cf, false, false);
      sf[f] = cf;
    }

    // drain prefetch into the other buffer now: the S-phase covered most of the
    // global latency, and the staged registers die before the softmax/PV phase
    if (it < 15) storeKVM(cur ^ 1);

    // ---- mask words: lane's 4 columns per row are 4 consecutive bytes ----
    unsigned mw[8];
#pragma unroll
    for (int j = 0; j < 8; ++j)
      mw[j] = *(const unsigned*)&Msk[cur][rW + hlf * 8 + j][nl * 4];

    // ---- row max over the unmasked SUPERSET (valid stabilizer; masked P
    //      entries are zeroed after exp below); DPP butterfly, no LDS traffic ----
    float rmax[8];
#pragma unroll
    for (int j = 0; j < 8; ++j) {
      float r = fmaxf(fmaxf(sf[0][j], sf[1][j]), fmaxf(sf[2][j], sf[3][j]));
      rmax[j] = rowMax16(r);
    }

    // ---- online softmax update (base-2 domain) ----
    float rsum[8];
#pragma unroll
    for (int j = 0; j < 8; ++j) {
      float mnew  = fmaxf(m[j], rmax[j]);
      float alpha = __builtin_amdgcn_exp2f(m[j] - mnew);
      m[j] = mnew;
      l[j] *= alpha;
#pragma unroll
      for (int f = 0; f < 4; ++f) acc[f][j] *= alpha;
      float s_ = 0.f;
#pragma unroll
      for (int f = 0; f < 4; ++f) {
        // (float)((mw >> 8f) & 0xff) -> single v_cvt_f32_ubyte{f}
        float mk = (float)((mw[j] >> (8 * f)) & 0xffu);
        float p  = __builtin_amdgcn_exp2f(sf[f][j] - mnew) * mk;
        sf[f][j] = p;
        s_ += p;
      }
      rsum[j] = s_;
    }
#pragma unroll
    for (int j = 0; j < 8; ++j) l[j] += rowSum16(rsum[j]);

    // ---- P -> LDS: lane's 4 columns are contiguous -> one b64 per row ----
#pragma unroll
    for (int j = 0; j < 8; ++j) {
      v2h plo = pkrtz(sf[0][j], sf[1][j]);
      v2h phi = pkrtz(sf[2][j], sf[3][j]);
      v4h pp  = __builtin_shufflevector(plo, phi, 0,1,2,3);
      *(v4h*)&Pl[rW + hlf * 8 + j][nl * 4] = pp;
    }

    const h16* prow = &Pl[rW + nl][0];
    v16h pa0 = ldA(prow, 0, kg);
    v16h pa1 = ldA(prow, 1, kg);

    // ---- O += P * V^T  (B[k=s, n=c] from Vl[c][s]; c columns unpermuted) ----
#pragma unroll
    for (int f = 0; f < 4; ++f) {
      const h16* vrow = &Vl[cur][f * 16 + nl][hlf * 8];
      v16h bv0 = ldB(vrow);
      v16h bv1 = ldB(vrow + 32);
      acc[f] = __builtin_amdgcn_wmma_f32_16x16x32_f16(false, pa0, false, bv0, (short)0, acc[f], false, false);
      acc[f] = __builtin_amdgcn_wmma_f32_16x16x32_f16(false, pa1, false, bv1, (short)0, acc[f], false, false);
    }

    __syncthreads();                               // one barrier per tile
  }

  // ---- epilogue: divide by row sum, write out[b, h*64+c, t] (t contiguous) ----
  float invl[8];
#pragma unroll
  for (int j = 0; j < 8; ++j)
    invl[j] = __builtin_amdgcn_rcpf(fmaxf(l[j], 1e-30f));

  const size_t ob = (size_t)b * (1024u * 1024u) + (size_t)hh * (64u * 1024u);
#pragma unroll
  for (int f = 0; f < 4; ++f) {
    float4 a0, a1;
    a0.x = acc[f][0] * invl[0]; a0.y = acc[f][1] * invl[1];
    a0.z = acc[f][2] * invl[2]; a0.w = acc[f][3] * invl[3];
    a1.x = acc[f][4] * invl[4]; a1.y = acc[f][5] * invl[5];
    a1.z = acc[f][6] * invl[6]; a1.w = acc[f][7] * invl[7];
    float* po = out + ob + (size_t)(f * 16 + nl) * T_LEN + t0 + rW + hlf * 8;
    *(float4*)(po)     = a0;
    *(float4*)(po + 4) = a1;
  }
}

extern "C" void kernel_launch(void* const* d_in, const int* in_sizes, int n_in,
                              void* d_out, int out_size, void* d_ws, size_t ws_size,
                              hipStream_t stream) {
  const float*         qkv     = (const float*)d_in[0];
  const unsigned char* mask    = (const unsigned char*)d_in[1];
  const float*         qk_bias = (const float*)d_in[2];
  float*               out     = (float*)d_out;
  (void)in_sizes; (void)n_in; (void)out_size; (void)d_ws; (void)ws_size;

  dim3 grid(16, 128);   // x: 16 query tiles of 64, y: 128 batch-heads
  dim3 block(128);      // 4 waves; each wave owns a 16-row query stripe
  hipLaunchKernelGGL(qkv_flash_attn, grid, block, 0, stream, qkv, mask, qk_bias, out);
}
